// ManyHeadAttention_70781061038126
// MI455X (gfx1250) — matrix operand order repaired and checked by reading, not compiled, not run on hardware
//
#include <hip/hip_runtime.h>
#include <hip/hip_bf16.h>

// ---------------------------------------------------------------------------
// MI455X (gfx1250) fused relative-position multi-head attention.
// f16 WMMA (v_wmma_f32_16x16x32_f16), fp32 accumulate, async-to-LDS staging.
// ---------------------------------------------------------------------------

typedef __attribute__((ext_vector_type(16))) _Float16 v16h;
typedef __attribute__((ext_vector_type(8)))  float    v8f;

#define UNITS 1024
#define HEADS 16
#define DEPTH 64
#define LSEQ  1024
#define BATCH 4
#define MREL  129
#define MPAD  160   // padded rel buckets (multiple of 32 for WMMA K-loop)

// ---- gfx1250 async copy: global -> LDS, 16B per lane, tracked by ASYNCcnt --
__device__ __forceinline__ void async_g2l_b128(unsigned lds_off, const void* gaddr) {
    asm volatile("global_load_async_to_lds_b128 %0, %1, off"
                 :: "v"(lds_off), "v"(gaddr) : "memory");
}
__device__ __forceinline__ void wait_async0() {
    asm volatile("s_wait_asynccnt 0" ::: "memory");
}
__device__ __forceinline__ unsigned lds_off(const void* p) {
    return (unsigned)(uintptr_t)p;   // LDS aperture: addr[31:0] == LDS byte offset
}

// ---- fragment helpers (per CDNA5 ISA 7.12.2 wave32 layouts) ----------------
// A (16x32 f16): lane l: row=l%16, kb=(l/16)*8; halves 0..7 -> K=kb..kb+7,
// halves 8..15 -> K=kb+16..kb+23. Caller passes p = base + row*ld + kb.
__device__ __forceinline__ v16h ldA(const _Float16* p) {
    v16h r;
#pragma unroll
    for (int i = 0; i < 8; ++i) { r[i] = p[i]; r[i + 8] = p[i + 16]; }
    return r;
}
// B (32x16 f16): lane l: col n=l%16, kb2=(l/16)*8; halves 0..7 -> K=kb2..kb2+7,
// halves 8..15 -> K=kb2+16..kb2+23. Caller passes p = base + n*ld + kb2.
__device__ __forceinline__ v16h ldB(const _Float16* p) {
    v16h r;
#pragma unroll
    for (int i = 0; i < 8; ++i) { r[i] = p[i]; r[i + 8] = p[i + 16]; }
    return r;
}
// A fragment sourced from fp32 LDS (converted on the fly).
__device__ __forceinline__ v16h ldAf32(const float* p) {
    v16h r;
#pragma unroll
    for (int i = 0; i < 8; ++i) {
        r[i]     = (_Float16)p[i];
        r[i + 8] = (_Float16)p[i + 16];
    }
    return r;
}
__device__ __forceinline__ v8f wmma16(v16h a, v16h b, v8f c) {
    return __builtin_amdgcn_wmma_f32_16x16x32_f16(
        false, a, false, b, (short)0, c, false, false);
}

// ---- prep kernels ----------------------------------------------------------
__global__ void cvt_f16_kernel(const float* __restrict__ s, _Float16* __restrict__ d, int n) {
    for (int i = blockIdx.x * blockDim.x + threadIdx.x; i < n; i += gridDim.x * blockDim.x)
        d[i] = (_Float16)s[i];
}

// w [K=1024][N=1024] -> wt [N][K] f16, LDS-tiled (coalesced both ways)
__global__ void transpose_cvt_kernel(const float* __restrict__ w, _Float16* __restrict__ wt) {
    __shared__ float tile[32][33];
    int tx = threadIdx.x & 31, ty = threadIdx.x >> 5;   // 32 x 8
    int kb = blockIdx.y * 32, nb = blockIdx.x * 32;
#pragma unroll
    for (int t = 0; t < 4; ++t)
        tile[ty + t * 8][tx] = w[(size_t)(kb + ty + t * 8) * 1024 + nb + tx];
    __syncthreads();
#pragma unroll
    for (int t = 0; t < 4; ++t)
        wt[(size_t)(nb + ty + t * 8) * 1024 + kb + tx] = (_Float16)tile[tx][ty + t * 8];
}

// ek [H][129][64] -> ekp [H][160][64] f16, zero padded
__global__ void prep_ek_kernel(const float* __restrict__ ek, _Float16* __restrict__ ekp) {
    int idx = blockIdx.x * blockDim.x + threadIdx.x;
    if (idx >= HEADS * MPAD * DEPTH) return;
    int d = idx & 63;
    int m = (idx >> 6) % MPAD;
    int h = idx / (MPAD * DEPTH);
    ekp[idx] = (m < MREL) ? (_Float16)ek[((size_t)h * MREL + m) * DEPTH + d] : (_Float16)0.0f;
}

// ev [H][129][64] -> evt [H][64][160] f16 (transposed, zero padded)
__global__ void prep_evt_kernel(const float* __restrict__ ev, _Float16* __restrict__ evt) {
    int idx = blockIdx.x * blockDim.x + threadIdx.x;
    if (idx >= HEADS * DEPTH * MPAD) return;
    int m = idx % MPAD;
    int d = (idx / MPAD) & 63;
    int h = idx / (DEPTH * MPAD);
    evt[idx] = (m < MREL) ? (_Float16)ev[((size_t)h * MREL + m) * DEPTH + d] : (_Float16)0.0f;
}

// ---- WMMA GEMM with double-buffered async-to-LDS staging -------------------
// C[M,N] = A[M,K] * BT[N,K]^T ; M=4096, N=K=1024.
// MODE 0: f16 out, head-split [B,H,L,D]            (Q, K)
// MODE 1: f16 out, head-split transposed [B,H,D,L] (V^T)
// MODE 2: fp32 out row-major + bias                (final projection)
// Block: 256 thr (8 waves, 4x2), tile 128(M) x 64(N), BK=64, 2 LDS buffers.
template <int MODE>
__global__ void __launch_bounds__(256)
gemm_wmma_kernel(const _Float16* __restrict__ A,
                 const _Float16* __restrict__ BT,
                 const float* __restrict__ bias,
                 void* __restrict__ Cout) {
    const int K = 1024, N = 1024;
    __shared__ _Float16 As[2][128][64];   // 32 KB
    __shared__ _Float16 Bs[2][64][64];    // 16 KB

    int tid = threadIdx.x, lane = tid & 31, wave = tid >> 5;
    int wm = wave & 3, wn = wave >> 2;
    int m0 = blockIdx.x * 128, n0 = blockIdx.y * 64;
    int mw = wm * 32, nw = wn * 32;
    int lr = lane & 15, lh = lane >> 4;

    auto stage = [&](int buf, int kk) {
        // A tile: 128 rows x 128B = 1024 x 16B chunks (4 per thread)
#pragma unroll
        for (int t = tid; t < 1024; t += 256) {
            int row = t >> 3, seg = (t & 7) * 8;
            async_g2l_b128(lds_off(&As[buf][row][seg]),
                           A + (size_t)(m0 + row) * K + kk + seg);
        }
        // B tile: 64 rows x 128B = 512 x 16B chunks (2 per thread)
#pragma unroll
        for (int t = tid; t < 512; t += 256) {
            int row = t >> 3, seg = (t & 7) * 8;
            async_g2l_b128(lds_off(&Bs[buf][row][seg]),
                           BT + (size_t)(n0 + row) * K + kk + seg);
        }
    };

    stage(0, 0);
    wait_async0();
    __syncthreads();

    v8f acc[2][2] = {};
    int buf = 0;
    for (int kk = 0; kk < K; kk += 64) {
        if (kk + 64 < K) stage(buf ^ 1, kk + 64);

#pragma unroll
        for (int ks = 0; ks < 2; ++ks) {
            int ka = ks * 32 + lh * 8, kb = ks * 32 + lh * 8;
            v16h a0 = ldA(&As[buf][mw + lr][ka]);
            v16h a1 = ldA(&As[buf][mw + 16 + lr][ka]);
            v16h b0 = ldB(&Bs[buf][nw + lr][kb]);
            v16h b1 = ldB(&Bs[buf][nw + 16 + lr][kb]);
            acc[0][0] = wmma16(a0, b0, acc[0][0]);
            acc[0][1] = wmma16(a0, b1, acc[0][1]);
            acc[1][0] = wmma16(a1, b0, acc[1][0]);
            acc[1][1] = wmma16(a1, b1, acc[1][1]);
        }

        if (kk + 64 < K) wait_async0();
        __syncthreads();
        buf ^= 1;
    }

    // Epilogue. Tile alignment => b and h are block-constant.
    int b = m0 >> 10;                  // 128-row tile never crosses a batch
    int h = blockIdx.y;                // 64-col tile == one head (N/64 == HEADS)
    int lbase = (m0 & 1023) + mw + lh * 8;
#pragma unroll
    for (int rt = 0; rt < 2; ++rt)
#pragma unroll
        for (int nt = 0; nt < 2; ++nt) {
            int d = nw + nt * 16 + lr;
#pragma unroll
            for (int r = 0; r < 8; ++r) {
                int l = lbase + rt * 16 + r;
                float v = acc[rt][nt][r];
                if constexpr (MODE == 2) {
                    ((float*)Cout)[(size_t)(b * LSEQ + l) * N + n0 + d] = v + bias[n0 + d];
                } else if constexpr (MODE == 0) {
                    ((_Float16*)Cout)[(((size_t)(b * HEADS + h)) * LSEQ + l) * DEPTH + d] =
                        (_Float16)v;
                } else {
                    ((_Float16*)Cout)[(((size_t)(b * HEADS + h)) * DEPTH + d) * LSEQ + l] =
                        (_Float16)v;
                }
            }
        }
}

// ---- fused attention kernel ------------------------------------------------
// One block = one (b,h) x 32 query rows. Full [32x1024] score block in LDS.
// Dynamic LDS: S f32[32][1024] | P f16[32][1024] | QM f32[32][160]
//              | WR f32[32][160] | RED f32[32][8] | Qs f16[32][64]
//              | EKs f16[160][64] | EVs f16[64][160]   (~277 KB total)
__global__ void __launch_bounds__(256)
attn_rel_kernel(const _Float16* __restrict__ Q,
                const _Float16* __restrict__ Kb,
                const _Float16* __restrict__ VT,
                const _Float16* __restrict__ ekp,
                const _Float16* __restrict__ evt,
                const float* __restrict__ pad_mask,
                _Float16* __restrict__ out) {
    extern __shared__ char smem[];
    float*    S   = (float*)smem;                                  // 131072 B
    _Float16* P   = (_Float16*)(smem + 131072);                    //  65536 B
    float*    QM  = (float*)(smem + 196608);                       //  20480 B
    float*    WR  = (float*)(smem + 217088);                       //  20480 B
    float*    RED = (float*)(smem + 237568);                       //   1024 B
    _Float16* Qs  = (_Float16*)(smem + 238592);                    //   4096 B
    _Float16* EKs = (_Float16*)(smem + 242688);                    //  20480 B
    _Float16* EVs = (_Float16*)(smem + 263168);                    //  20480 B

    int tid = threadIdx.x, lane = tid & 31, wave = tid >> 5;
    int lr = lane & 15, lh = lane >> 4;
    int bh = blockIdx.y;                 // b*16 + h
    int b = bh >> 4, h = bh & 15;
    int i0 = blockIdx.x * 32;

    const _Float16* Qh  = Q  + ((size_t)bh * LSEQ + i0) * DEPTH;   // [32][64]
    const _Float16* Kh  = Kb + (size_t)bh * LSEQ * DEPTH;          // [1024][64]
    const _Float16* Vh  = VT + (size_t)bh * DEPTH * LSEQ;          // [64][1024]
    const _Float16* ekh = ekp + (size_t)h * MPAD * DEPTH;          // [160][64]
    const _Float16* evh = evt + (size_t)h * DEPTH * MPAD;          // [64][160]

    // async-stage reused operands into LDS (Q tile, ek, ev^T)
    async_g2l_b128(lds_off(Qs + tid * 8), Qh + tid * 8);           // 4 KB
#pragma unroll
    for (int t = tid; t < 1280; t += 256) {                        // 20 KB each
        async_g2l_b128(lds_off(EKs + t * 8), ekh + t * 8);
        async_g2l_b128(lds_off(EVs + t * 8), evh + t * 8);
    }
    for (int idx = tid; idx < 32 * MPAD; idx += 256) WR[idx] = 0.0f;
    wait_async0();
    __syncthreads();

    // hoist the 4 Q A-fragments (reused by every qm/S fragment); constant-indexed
    v16h qa00 = ldA(Qs + (size_t)lr * DEPTH + lh * 8);
    v16h qa01 = ldA(Qs + (size_t)lr * DEPTH + 32 + lh * 8);
    v16h qa10 = ldA(Qs + (size_t)(16 + lr) * DEPTH + lh * 8);
    v16h qa11 = ldA(Qs + (size_t)(16 + lr) * DEPTH + 32 + lh * 8);

    // --- QM = q @ ek^T : 10 mtiles over 8 waves, both rowtiles per tile ---
    for (int mt = wave; mt < 10; mt += 8) {
        v16h b0 = ldB(EKs + (size_t)(mt * 16 + lr) * DEPTH + lh * 8);
        v16h b1 = ldB(EKs + (size_t)(mt * 16 + lr) * DEPTH + 32 + lh * 8);
        v8f a0 = {}, a1 = {};
        a0 = wmma16(qa00, b0, a0);
        a0 = wmma16(qa01, b1, a0);
        a1 = wmma16(qa10, b0, a1);
        a1 = wmma16(qa11, b1, a1);
#pragma unroll
        for (int r = 0; r < 8; ++r) {
            QM[(lh * 8 + r) * MPAD + mt * 16 + lr]        = a0[r];
            QM[(16 + lh * 8 + r) * MPAD + mt * 16 + lr]   = a1[r];
        }
    }

    // --- S = q @ k^T : 64 jtiles over 8 waves, both rowtiles per tile ---
#pragma unroll 2
    for (int it = 0; it < 8; ++it) {
        int jt = it * 8 + wave;
        __builtin_prefetch(Kh + (size_t)(((it + 1) * 8 + wave) & 63) * 16 * DEPTH, 0, 3);
        v16h b0 = ldB(Kh + (size_t)(jt * 16 + lr) * DEPTH + lh * 8);
        v16h b1 = ldB(Kh + (size_t)(jt * 16 + lr) * DEPTH + 32 + lh * 8);
        v8f a0 = {}, a1 = {};
        a0 = wmma16(qa00, b0, a0);
        a0 = wmma16(qa01, b1, a0);
        a1 = wmma16(qa10, b0, a1);
        a1 = wmma16(qa11, b1, a1);
#pragma unroll
        for (int r = 0; r < 8; ++r) {
            S[(lh * 8 + r) * LSEQ + jt * 16 + lr]      = a0[r];
            S[(16 + lh * 8 + r) * LSEQ + jt * 16 + lr] = a1[r];
        }
    }
    __syncthreads();

    // --- exact softmax + rel-k gather + rel-v bucket scatter ---------------
    // 8 threads per row, 128 contiguous columns each.
    {
        int r = tid >> 3, c = tid & 7;
        int i = i0 + r;
        int j0 = c * 128, j1 = j0 + 128;
        const float* mrow = pad_mask + (size_t)b * LSEQ;
        float*       srow = S + (size_t)r * LSEQ;
        const float* qmr  = QM + (size_t)r * MPAD;
        float*       wrr  = WR + (size_t)r * MPAD;
        _Float16*    prow = P + (size_t)r * LSEQ;

        float mx = -1e30f;
        for (int j = j0; j < j1; ++j) {
            int off = j - i;
            off = off < -64 ? -64 : (off > 64 ? 64 : off);
            float val = (srow[j] + qmr[off + 64]) * 0.125f + mrow[j] * -1e9f;
            srow[j] = val;
            mx = fmaxf(mx, val);
        }
        RED[r * 8 + c] = mx;
        __syncthreads();
#pragma unroll
        for (int t = 0; t < 8; ++t) mx = fmaxf(mx, RED[r * 8 + t]);

        float sum = 0.0f;
        for (int j = j0; j < j1; ++j) {
            float e = __expf(srow[j] - mx);
            srow[j] = e;
            sum += e;
        }
        __syncthreads();                 // RED reuse
        RED[r * 8 + c] = sum;
        __syncthreads();
        sum = 0.0f;
#pragma unroll
        for (int t = 0; t < 8; ++t) sum += RED[r * 8 + t];
        float inv = 1.0f / sum;

        float w0 = 0.0f, w128 = 0.0f;    // clamped buckets (shared across chunks)
        for (int j = j0; j < j1; ++j) {
            float p = srow[j] * inv;
            prow[j] = (_Float16)p;
            int off = j - i;
            if (off <= -64)      w0   += p;
            else if (off >= 64)  w128 += p;
            else                 wrr[off + 64] += p;   // unique owner thread
        }
        if (w0   != 0.0f) atomicAdd(&wrr[0],   w0);    // ds_add_f32
        if (w128 != 0.0f) atomicAdd(&wrr[128], w128);
    }
    __syncthreads();

    // --- O = P @ V + WR @ ev^T : 2 rowtiles x 4 dtiles = 8 frags, 1/wave ---
    {
        int rt = wave >> 2, dt = wave & 3;
        v8f acc = {};
#pragma unroll 4
        for (int ks = 0; ks < 32; ++ks) {        // K loop over j (1024)
            v16h a  = ldA(P + (size_t)(rt * 16 + lr) * LSEQ + ks * 32 + lh * 8);
            v16h bb = ldB(Vh + (size_t)(dt * 16 + lr) * LSEQ + ks * 32 + lh * 8);
            acc = wmma16(a, bb, acc);
        }
#pragma unroll
        for (int ks = 0; ks < 5; ++ks) {         // K loop over rel buckets (160)
            v16h a  = ldAf32(WR + (size_t)(rt * 16 + lr) * MPAD + ks * 32 + lh * 8);
            v16h bb = ldB(EVs + (size_t)(dt * 16 + lr) * MPAD + ks * 32 + lh * 8);
            acc = wmma16(a, bb, acc);
        }
#pragma unroll
        for (int r = 0; r < 8; ++r) {            // [B][L][H*D] f16
            int row = rt * 16 + lh * 8 + r;
            int d   = dt * 16 + lr;
            out[((size_t)b * LSEQ + i0 + row) * UNITS + h * DEPTH + d] = (_Float16)acc[r];
        }
    }
}

// ---------------------------------------------------------------------------
extern "C" void kernel_launch(void* const* d_in, const int* in_sizes, int n_in,
                              void* d_out, int out_size, void* d_ws, size_t ws_size,
                              hipStream_t stream) {
    (void)in_sizes; (void)n_in; (void)out_size; (void)ws_size;
    const float* x   = (const float*)d_in[0];
    const float* pad = (const float*)d_in[1];
    const float* wq  = (const float*)d_in[2];
    const float* wk  = (const float*)d_in[3];
    const float* wv  = (const float*)d_in[4];
    const float* wo  = (const float*)d_in[5];
    const float* bo  = (const float*)d_in[6];
    const float* ek  = (const float*)d_in[7];
    const float* ev  = (const float*)d_in[8];

    const size_t nBLU = (size_t)BATCH * LSEQ * UNITS;   // 4,194,304
    const size_t nW   = (size_t)UNITS * UNITS;          // 1,048,576
    const size_t nEK  = (size_t)HEADS * MPAD * DEPTH;   // 163,840

    _Float16* ws = (_Float16*)d_ws;
    size_t o = 0;
    _Float16* xh   = ws + o; o += nBLU;
    _Float16* wqT  = ws + o; o += nW;
    _Float16* wkT  = ws + o; o += nW;
    _Float16* wvT  = ws + o; o += nW;
    _Float16* woT  = ws + o; o += nW;
    _Float16* Qb   = ws + o; o += nBLU;
    _Float16* Kb   = ws + o; o += nBLU;
    _Float16* VTb  = ws + o; o += nBLU;
    _Float16* ekp  = ws + o; o += nEK;
    _Float16* evtp = ws + o; o += nEK;
    _Float16* attn = ws + o; o += nBLU;

    cvt_f16_kernel<<<2048, 256, 0, stream>>>(x, xh, (int)nBLU);
    dim3 tg(32, 32);
    transpose_cvt_kernel<<<tg, 256, 0, stream>>>(wq, wqT);
    transpose_cvt_kernel<<<tg, 256, 0, stream>>>(wk, wkT);
    transpose_cvt_kernel<<<tg, 256, 0, stream>>>(wv, wvT);
    transpose_cvt_kernel<<<tg, 256, 0, stream>>>(wo, woT);
    prep_ek_kernel<<<640, 256, 0, stream>>>(ek, ekp);
    prep_evt_kernel<<<640, 256, 0, stream>>>(ev, evtp);

    dim3 gg(32, 16);
    gemm_wmma_kernel<0><<<gg, 256, 0, stream>>>(xh, wqT, nullptr, Qb);
    gemm_wmma_kernel<0><<<gg, 256, 0, stream>>>(xh, wkT, nullptr, Kb);
    gemm_wmma_kernel<1><<<gg, 256, 0, stream>>>(xh, wvT, nullptr, VTb);

    const size_t smem = 283648;          // see layout above
    attn_rel_kernel<<<dim3(32, 64), 256, smem, stream>>>(Qb, Kb, VTb, ekp, evtp, pad, attn);

    gemm_wmma_kernel<2><<<gg, 256, 0, stream>>>(attn, woT, bo, d_out);
}
